// AttnEncoder_25177098289638
// MI455X (gfx1250) — matrix operand order repaired and checked by reading, not compiled, not run on hardware
//
#include <hip/hip_runtime.h>
#include <hip/hip_fp16.h>

typedef __attribute__((ext_vector_type(16))) _Float16 v16h;
typedef __attribute__((ext_vector_type(8)))  _Float16 v8h;
typedef __attribute__((ext_vector_type(4)))  _Float16 v4h;
typedef __attribute__((ext_vector_type(8)))  float    v8f;
typedef __attribute__((ext_vector_type(4)))  float    v4f;

static constexpr int Bb = 1024, T = 128, N = 256, H = 128, G = 512; // G = 4H

// workspace layout (bytes)
static constexpr size_t WS_WIH  = 0;                         // 512*256 f16
static constexpr size_t WS_WHH  = WS_WIH + (size_t)G*N*2;    // 512*128 f16
static constexpr size_t WS_BIAS = WS_WHH + (size_t)G*H*2;    // 512 f32
static constexpr size_t WS_A    = WS_BIAS + (size_t)G*4;     // 1024*256 f32

// ---------------------------------------------------------------------------
// Prepass 1: convert weights to f16 (layout-preserving: W[col][k] is already
// the WMMA-B access pattern), fold the two bias vectors together.
// ---------------------------------------------------------------------------
__global__ void k_prep_weights(const float* __restrict__ w_ih,
                               const float* __restrict__ w_hh,
                               const float* __restrict__ b_ih,
                               const float* __restrict__ b_hh,
                               __half* __restrict__ wih16,
                               __half* __restrict__ whh16,
                               float* __restrict__ bias)
{
    const int row = blockIdx.x;      // 0..511 (gate output index)
    const int tid = threadIdx.x;     // 0..255
    wih16[row * N + tid] = __float2half(w_ih[row * N + tid]);
    if (tid < H) whh16[row * H + tid] = __float2half(w_hh[row * H + tid]);
    if (tid == 0) bias[row] = b_ih[row] + b_hh[row];
}

// ---------------------------------------------------------------------------
// Prepass 2: feat[b,n] = sum_t x[b,t,n]*wx[t] + attn_b ; a = softmax_n(feat).
// The recurrent logit term (h@wh + c@wc) is a per-row constant, which softmax
// is invariant to -> the attention matrix is time-invariant: compute it once.
// ---------------------------------------------------------------------------
__global__ void k_attn(const float* __restrict__ x,       // (B,T,N)
                       const float* __restrict__ attn_w,  // (1, 2H+T)
                       const float* __restrict__ attn_b,  // (1,)
                       float* __restrict__ a_out)         // (B,N)
{
    __shared__ float wx[T];
    __shared__ float red[8];
    const int b = blockIdx.x, n = threadIdx.x;
    if (n < T) wx[n] = attn_w[2 * H + n];
    __syncthreads();

    const float* xp = x + (size_t)b * T * N + n;
    float s = attn_b[0];
#pragma unroll 4
    for (int t = 0; t < T; ++t) s += xp[(size_t)t * N] * wx[t];

    // block softmax over the 256 values (one per thread)
    float m = s;
    for (int off = 16; off; off >>= 1) m = fmaxf(m, __shfl_xor(m, off, 32));
    if ((n & 31) == 0) red[n >> 5] = m;
    __syncthreads();
    m = red[0];
#pragma unroll
    for (int i = 1; i < 8; ++i) m = fmaxf(m, red[i]);
    __syncthreads();

    const float e = __expf(s - m);
    float sum = e;
    for (int off = 16; off; off >>= 1) sum += __shfl_xor(sum, off, 32);
    if ((n & 31) == 0) red[n >> 5] = sum;
    __syncthreads();
    sum = 0.f;
#pragma unroll
    for (int i = 0; i < 8; ++i) sum += red[i];
    a_out[(size_t)b * N + n] = e / sum;
}

// ---------------------------------------------------------------------------
// Main kernel: 64 blocks x 512 threads (16 waves). Block owns 16 batch rows
// and runs the full T=128 recurrence. Each wave produces a 16x32 gate slice:
// 2 N-tiles x 12 K-chunks = 24 v_wmma_f32_16x16x32_f16 per step. The 24
// loop-invariant B-fragments (f16 weights) fit in <256 VGPRs together with
// the accumulators, so they stay register-resident across the whole time
// loop without s_set_vgpr_msb high-register addressing.
//
// A-fragment LDS layout: each 32-K chunk is stored in natural k order,
// so lane-group g (lane>>4) reads its 16 halves from one contiguous 32B run
// (contiguous k order, same as the B-fragments): k(h) = g*16 + h.
// B-fragments: lanes 0-15 hold K=k0..k0+15, lanes 16-31 hold K=k0+16..k0+31,
// contiguous in the row-major f16 weight rows.
// ---------------------------------------------------------------------------
__device__ __forceinline__ float sigmf(float x) { return 1.f / (1.f + __expf(-x)); }
__device__ __forceinline__ float tanhfast(float x) {
    float e = __expf(2.f * x);
    return (e - 1.f) / (e + 1.f);
}
__device__ __forceinline__ int kperm(int p) {   // 8-group start within 32-chunk
    return p;
}

__global__ void __launch_bounds__(512)
k_encoder(const float*  __restrict__ x,       // (B,T,N)
          const __half* __restrict__ wih16,   // (G,N) f16
          const __half* __restrict__ whh16,   // (G,H) f16
          const float*  __restrict__ bias,    // (G,)
          const float*  __restrict__ a_in,    // (B,N)
          const float*  __restrict__ h0,      // (B,H)
          const float*  __restrict__ c0,      // (B,H)
          float* __restrict__ attn_out,       // (B,T,N)
          float* __restrict__ enc_out)        // (B,T,H)
{
    __shared__ __align__(32) _Float16 s_wi[16 * N];  // 8 KB
    __shared__ __align__(32) _Float16 s_h [16 * H];  // 4 KB
    __shared__ float s_c[16 * H];                    // 8 KB
    __shared__ float s_g[16 * G];                    // 32 KB
    __shared__ float s_b[G];                         // 2 KB

    const int tid = threadIdx.x;     // 0..511
    const int b0  = blockIdx.x * 16;
    const int row = tid >> 5;        // 0..15 : batch row within tile
    const int c8  = tid & 31;        // 0..31 : 8-col group within the row

    s_b[tid] = bias[tid];

    // attention weights for this thread's 8 columns: constant over time
    v4f a4[2];
    {
        const float* ap = a_in + (size_t)(b0 + row) * N + c8 * 8;
        a4[0] = *(const v4f*)ap;
        a4[1] = *(const v4f*)(ap + 4);
    }
    // init c (f32) and h (f16) : 4 hidden units per thread
    {
        const int j0 = (tid & 31) * 4;               // 0..124
        const float* cp = c0 + (size_t)(b0 + row) * H + j0;
        const float* hp = h0 + (size_t)(b0 + row) * H + j0;
        v4h hh;
#pragma unroll
        for (int m = 0; m < 4; ++m) {
            s_c[row * H + j0 + m] = cp[m];
            hh[m] = (_Float16)hp[m];
        }
        const int pc = j0 & 31;
        *(v4h*)&s_h[row * H + (j0 >> 5) * 32 + kperm(pc & 24) + (pc & 7)] = hh;
    }
    __syncthreads();

    const int wave  = tid >> 5, lane = tid & 31;
    const int r16   = lane & 15, gsel = lane >> 4;
    const int ncol0 = wave * 32;                 // 32 gate columns per wave

#pragma unroll 1
    for (int t = 0; t < T; ++t) {
        // -------- phase 1: wi = a*x_t -> LDS (f16); stream a out
        {
            const float* xp = x + ((size_t)(b0 + row) * T + t) * N + c8 * 8;
            v4f x0 = *(const v4f*)xp;
            v4f x1 = *(const v4f*)(xp + 4);
            if (t + 1 < T) __builtin_prefetch(xp + N, 0, 1);

            float* aop = attn_out + ((size_t)(b0 + row) * T + t) * N + c8 * 8;
            *(v4f*)aop       = a4[0];
            *(v4f*)(aop + 4) = a4[1];

            v8h wi;
#pragma unroll
            for (int e = 0; e < 4; ++e) {
                wi[e]     = (_Float16)(x0[e] * a4[0][e]);
                wi[e + 4] = (_Float16)(x1[e] * a4[1][e]);
            }
            const int pc = (c8 * 8) & 31;                   // 0,8,16,24
            *(v8h*)&s_wi[row * N + (c8 >> 2) * 32 + kperm(pc)] = wi;
        }
        __syncthreads();

        // -------- phase 2: gates = wi@Wih^T + h@Whh^T + bias   (WMMA)
        v8f acc[2];
#pragma unroll
        for (int j = 0; j < 2; ++j) {
            const float bv = s_b[ncol0 + j * 16 + r16];
#pragma unroll
            for (int v = 0; v < 8; ++v) acc[j][v] = bv;
        }
#pragma unroll
        for (int k0 = 0; k0 < N; k0 += 32) {
            union { v16h v; v8h h[2]; } A;
            A.h[0] = *(const v8h*)&s_wi[r16 * N + k0 + gsel * 16];
            A.h[1] = *(const v8h*)&s_wi[r16 * N + k0 + gsel * 16 + 8];
#pragma unroll
            for (int j = 0; j < 2; ++j) {
                const __half* bp = wih16 + (size_t)(ncol0 + j * 16 + r16) * N + k0 + gsel * 16;
                union { v16h v; v8h h[2]; } Bm;
                Bm.h[0] = *(const v8h*)(const void*)bp;
                Bm.h[1] = *(const v8h*)(const void*)(bp + 8);
                acc[j] = __builtin_amdgcn_wmma_f32_16x16x32_f16(
                    false, A.v, false, Bm.v, (short)0, acc[j], false, false);
            }
        }
#pragma unroll
        for (int k0 = 0; k0 < H; k0 += 32) {
            union { v16h v; v8h h[2]; } A;
            A.h[0] = *(const v8h*)&s_h[r16 * H + k0 + gsel * 16];
            A.h[1] = *(const v8h*)&s_h[r16 * H + k0 + gsel * 16 + 8];
#pragma unroll
            for (int j = 0; j < 2; ++j) {
                const __half* bp = whh16 + (size_t)(ncol0 + j * 16 + r16) * H + k0 + gsel * 16;
                union { v16h v; v8h h[2]; } Bm;
                Bm.h[0] = *(const v8h*)(const void*)bp;
                Bm.h[1] = *(const v8h*)(const void*)(bp + 8);
                acc[j] = __builtin_amdgcn_wmma_f32_16x16x32_f16(
                    false, A.v, false, Bm.v, (short)0, acc[j], false, false);
            }
        }
        // -------- phase 3: spill gate tile (C/D layout: row = v + 8*gsel)
#pragma unroll
        for (int j = 0; j < 2; ++j)
#pragma unroll
            for (int v = 0; v < 8; ++v)
                s_g[(v + gsel * 8) * G + ncol0 + j * 16 + r16] = acc[j][v];
        __syncthreads();

        // -------- phase 4: LSTM pointwise; update c (f32) / h (f16), emit h
        {
            const int j0 = (tid & 31) * 4;               // 0..124
            v4f hv;
            v4h hh;
#pragma unroll
            for (int m = 0; m < 4; ++m) {
                const int j = j0 + m;
                const float iv = sigmf(s_g[row * G + j]);
                const float fv = sigmf(s_g[row * G + H + j]);
                const float gv = tanhfast(s_g[row * G + 2 * H + j]);
                const float ov = sigmf(s_g[row * G + 3 * H + j]);
                const float cn = fv * s_c[row * H + j] + iv * gv;
                s_c[row * H + j] = cn;
                const float hn = ov * tanhfast(cn);
                hv[m] = hn;
                hh[m] = (_Float16)hn;
            }
            float* ep = enc_out + ((size_t)(b0 + row) * T + t) * H + j0;
            *(v4f*)ep = hv;
            const int pc = j0 & 31;
            *(v4h*)&s_h[row * H + (j0 >> 5) * 32 + kperm(pc & 24) + (pc & 7)] = hh;
        }
        __syncthreads();
    }
}

// ---------------------------------------------------------------------------
extern "C" void kernel_launch(void* const* d_in, const int* in_sizes, int n_in,
                              void* d_out, int out_size, void* d_ws, size_t ws_size,
                              hipStream_t stream)
{
    const float* input  = (const float*)d_in[0];  // (B,T,N)
    const float* attn_w = (const float*)d_in[1];  // (1, 2H+T)
    const float* attn_b = (const float*)d_in[2];  // (1,)
    const float* w_ih   = (const float*)d_in[3];  // (4H,N)
    const float* w_hh   = (const float*)d_in[4];  // (4H,H)
    const float* b_ih   = (const float*)d_in[5];  // (4H,)
    const float* b_hh   = (const float*)d_in[6];  // (4H,)
    const float* h0     = (const float*)d_in[7];  // (B,H)
    const float* c0     = (const float*)d_in[8];  // (B,H)
    // d_in[9] = sample (unused)

    char*   ws    = (char*)d_ws;
    __half* wih16 = (__half*)(ws + WS_WIH);
    __half* whh16 = (__half*)(ws + WS_WHH);
    float*  biasw = (float*)(ws + WS_BIAS);
    float*  a_ws  = (float*)(ws + WS_A);

    float* attn_out = (float*)d_out;                       // (B,T,N)
    float* enc_out  = attn_out + (size_t)Bb * T * N;       // (B,T,H)

    k_prep_weights<<<G, 256, 0, stream>>>(w_ih, w_hh, b_ih, b_hh, wih16, whh16, biasw);
    k_attn<<<Bb, 256, 0, stream>>>(input, attn_w, attn_b, a_ws);
    k_encoder<<<Bb / 16, 512, 0, stream>>>(input, wih16, whh16, biasw, a_ws,
                                           h0, c0, attn_out, enc_out);
}
